// DCNv2Module_60275571032619
// MI455X (gfx1250) — matrix operand order repaired and checked by reading, not compiled, not run on hardware
//
#include <hip/hip_runtime.h>
#include <hip/hip_bf16.h>

// DCNv2 fused implicit-GEMM for MI455X (gfx1250, wave32, WMMA).
// B=8, C_IN=64, C_OUT=64, H=W=128, K=9 (3x3, pad 1), Kdim = C_IN*K = 576.

typedef __attribute__((ext_vector_type(16))) _Float16 v16h;
typedef __attribute__((ext_vector_type(8)))  _Float16 v8h;
typedef __attribute__((ext_vector_type(8)))  float    v8f;

#define NB        8
#define CIN       64
#define COUT      64
#define HH        128
#define WW        128
#define HWSZ      (HH * WW)          // 16384
#define KPTS      9
#define KDIM      (CIN * KPTS)       // 576
#define KSTEPS    (KDIM / 32)        // 18 WMMA K-steps
#define MTILE     32                 // pixels per workgroup
#define ASTRIDE   584                // padded LDS row stride in halves (multiple of 8)

// ---------------------------------------------------------------------------
// Pack weight[COUT][CIN][3][3] (f32) into exact CDNA5 B-fragment layout (f16):
//   bp[((s*4 + nt)*32 + lane)*16 + e] = W_B[K = s*32 + (lane>>4)*8 + (e&7) + (e>>3)*16]
//                                          [N = nt*16 + (lane&15)]
// with W_B[kdim][n] = weight[n][c][kp], kdim = kp*64 + c.
// Total: 18*4*32*16 = 36864 halves = 73728 bytes in d_ws.
// ---------------------------------------------------------------------------
__global__ __launch_bounds__(256) void dcn_pack_weights(
    const float* __restrict__ w, _Float16* __restrict__ bp) {
  int idx  = blockIdx.x * 256 + threadIdx.x;          // [0, 36864)
  int e    = idx & 15;
  int lane = (idx >> 4) & 31;
  int nt   = (idx >> 9) & 3;
  int s    = idx >> 11;
  int n    = nt * 16 + (lane & 15);
  int K    = s * 32 + ((lane >> 4) << 3) + (e & 7) + ((e >> 3) << 4);         // kdim in [0, 576)
  int kp   = K >> 6;                                  // kernel point 0..8
  int c    = K & 63;                                  // input channel
  bp[idx] = (_Float16)w[(n * CIN + c) * KPTS + kp];
}

// ---------------------------------------------------------------------------
// Main fused kernel: 4096 workgroups x 256 threads (8 wave32).
// ---------------------------------------------------------------------------
__global__ __launch_bounds__(256) void dcn_v2_wmma(
    const float* __restrict__ in,       // [B][CIN][H][W]
    const float* __restrict__ offm,     // [B][3K][H][W]
    const _Float16* __restrict__ bpack, // packed B fragments (d_ws)
    const float* __restrict__ bias,     // [COUT]
    float* __restrict__ out) {          // [B][COUT][H][W]
  __shared__ _Float16 Alds[MTILE * ASTRIDE];          // 37376 B, f16 A-tile
  __shared__ float    Olds[COUT * (MTILE + 1)];       //  8448 B, store staging

  const int tid = threadIdx.x;
  const int g0  = blockIdx.x * MTILE;                 // global pixel base
  const int b   = g0 >> 14;                           // HWSZ = 16384, 32 | HWSZ
  const int hwb = g0 & (HWSZ - 1);

  // ---------------- Phase 1: deformable bilinear sampling -> LDS A ---------
  {
    const int grp = tid >> 4;                         // 16 groups of 16 lanes
    const int ln  = tid & 15;
    const float* ombase = offm + (size_t)b * (3 * KPTS) * HWSZ;
    const float* inbase = in + (size_t)b * CIN * HWSZ;
    for (int pair = grp; pair < MTILE * KPTS; pair += 16) {
      const int p  = pair / KPTS;
      const int k  = pair - p * KPTS;
      const int hw = hwb + p;
      const int h  = hw >> 7;
      const int w  = hw & (WW - 1);
      const float dy = ombase[(size_t)(2 * k)     * HWSZ + hw];
      const float dx = ombase[(size_t)(2 * k + 1) * HWSZ + hw];
      const float ml = ombase[(size_t)(2 * KPTS + k) * HWSZ + hw];
      const float msk = 1.0f / (1.0f + __expf(-ml));
      const float ys = (float)(h + k / 3 - 1) + dy;
      const float xs = (float)(w + k % 3 - 1) + dx;
      const float y0f = floorf(ys), x0f = floorf(xs);
      const int   y0  = (int)y0f,   x0  = (int)x0f;
      const float wy  = ys - y0f,   wx  = xs - x0f;
      const int   y1  = y0 + 1,     x1  = x0 + 1;
      const bool vy0 = (y0 >= 0) & (y0 < HH), vy1 = (y1 >= 0) & (y1 < HH);
      const bool vx0 = (x0 >= 0) & (x0 < WW), vx1 = (x1 >= 0) & (x1 < WW);
      const int y0c = min(max(y0, 0), HH - 1), y1c = min(max(y1, 0), HH - 1);
      const int x0c = min(max(x0, 0), WW - 1), x1c = min(max(x1, 0), WW - 1);
      const float f00 = (vy0 & vx0) ? (1.f - wy) * (1.f - wx) * msk : 0.f;
      const float f01 = (vy0 & vx1) ? (1.f - wy) * wx         * msk : 0.f;
      const float f10 = (vy1 & vx0) ? wy * (1.f - wx)         * msk : 0.f;
      const float f11 = (vy1 & vx1) ? wy * wx                 * msk : 0.f;
      const int o00 = y0c * WW + x0c, o01 = y0c * WW + x1c;
      const int o10 = y1c * WW + x0c, o11 = y1c * WW + x1c;
      #pragma unroll
      for (int ci = 0; ci < 4; ++ci) {
        const int c = ln + ci * 16;
        const float* pc = inbase + (size_t)c * HWSZ;  // gathers hit 192MB L2
        const float v = pc[o00] * f00 + pc[o01] * f01 +
                        pc[o10] * f10 + pc[o11] * f11;
        Alds[p * ASTRIDE + k * CIN + c] = (_Float16)v;
      }
    }
  }
  __syncthreads();

  // ---------------- Phase 2: WMMA contraction (8 waves = 2 M x 4 N) --------
  {
    const int wv    = tid >> 5;
    const int lane  = tid & 31;
    const int mt    = wv >> 2;                        // 0..1
    const int nt    = wv & 3;                         // 0..3
    const int mbase = mt * 16;
    const int m     = mbase + (lane & 15);
    const int kgrp  = (lane >> 4) << 3;               // A frag K base: 0 or 8
    v8f acc = {};
    #pragma unroll
    for (int s = 0; s < KSTEPS; ++s) {
      const _Float16* ar = &Alds[m * ASTRIDE + s * 32 + kgrp];
      const v8h alo = *(const v8h*)(ar);              // K = base + 0..7
      const v8h ahi = *(const v8h*)(ar + 16);         // K = base + 16..23
      const v16h a = __builtin_shufflevector(alo, ahi,
          0, 1, 2, 3, 4, 5, 6, 7, 8, 9, 10, 11, 12, 13, 14, 15);
      const _Float16* bptr = bpack + (size_t)(((s * 4 + nt) * 32 + lane) * 16);
      if (s + 1 < KSTEPS)                             // global_prefetch_b8
        __builtin_prefetch(bptr + 4 * 32 * 16, 0, 0);
      const v16h bf = *(const v16h*)(bptr);
      acc = __builtin_amdgcn_wmma_f32_16x16x32_f16(
          false, a, false, bf, (short)0, acc, false, false);
    }
    // bias + stage 16x16 f32 tile to LDS (C/D layout: VGPR i -> M = i + 8*(lane>>4))
    const int o  = nt * 16 + (lane & 15);
    const float bz = bias[o];
    #pragma unroll
    for (int i = 0; i < 8; ++i) {
      const int prow = mbase + ((lane >> 4) << 3) + i;
      Olds[o * (MTILE + 1) + prow] = acc[i] + bz;
    }
  }
  __syncthreads();

  // ---------------- Phase 3: coalesced NCHW store --------------------------
  {
    float* ob = out + (size_t)b * COUT * HWSZ + hwb;
    #pragma unroll
    for (int j = 0; j < 8; ++j) {
      const int idx = tid + j * 256;                  // 0..2047
      const int o   = idx >> 5;
      const int p   = idx & 31;                       // consecutive lanes -> consecutive w
      ob[(size_t)o * HWSZ + p] = Olds[o * (MTILE + 1) + p];
    }
  }
}

// ---------------------------------------------------------------------------
extern "C" void kernel_launch(void* const* d_in, const int* in_sizes, int n_in,
                              void* d_out, int out_size, void* d_ws, size_t ws_size,
                              hipStream_t stream) {
  const float* input  = (const float*)d_in[0];   // [8][64][128][128]
  const float* offm   = (const float*)d_in[1];   // [8][27][128][128]
  const float* weight = (const float*)d_in[2];   // [64][64][3][3]
  const float* bias   = (const float*)d_in[3];   // [64]
  float*       out    = (float*)d_out;           // [8][64][128][128]
  _Float16*    bpack  = (_Float16*)d_ws;         // 73728 bytes used

  // Pack weights into WMMA B-fragment layout (36864 halves / 256 = 144 blocks).
  dcn_pack_weights<<<144, 256, 0, stream>>>(weight, bpack);
  // Fused sample + implicit GEMM: (B*H*W)/MTILE = 131072/32 = 4096 workgroups.
  dcn_v2_wmma<<<4096, 256, 0, stream>>>(input, offm, bpack, bias, out);
}
